// Head_9131100471327
// MI455X (gfx1250) — matrix operand order repaired and checked by reading, not compiled, not run on hardware
//
#include <hip/hip_runtime.h>
#include <hip/hip_bf16.h>

#define B_DIM 8
#define T_DIM 2048
#define C_DIM 1024
#define H_DIM 64

#define KC     64                      // K-chunk staged in LDS per iteration
#define CHUNKS (C_DIM / KC)            // 16
#define LROWS  (3 * H_DIM)             // 192 weight rows (q,k,v)
#define LSTRIDE 72                     // f16 per LDS row (64 + 8 pad, 16B aligned)

typedef __attribute__((ext_vector_type(16))) _Float16 v16h;
typedef __attribute__((ext_vector_type(8)))  _Float16 v8h;
typedef __attribute__((ext_vector_type(8)))  float    v8f;
typedef __attribute__((ext_vector_type(4)))  float    f32x4;

// ---- fragment loaders (CDNA5 WMMA 16x16x32 f16 layouts, ISA 7.12.2) ----
// A-matrix 16x32: lanes 0-15 = rows, K(half0)=0..7 & 16..23, K(half1)=8..15 & 24..31
__device__ __forceinline__ v16h load_a_frag_f16(const _Float16* row, int kbase, int half) {
  const v8h lo = *(const v8h*)(row + kbase + half * 8);
  const v8h hi = *(const v8h*)(row + kbase + 16 + half * 8);
  return __builtin_shufflevector(lo, hi, 0,1,2,3,4,5,6,7,8,9,10,11,12,13,14,15);
}
// B-matrix 32x16: lane n = column, K(half0)=0..7 & 16..23, K(half1)=8..15 & 24..31
__device__ __forceinline__ v16h load_b_frag_f16(const _Float16* colrow, int kbase, int half) {
  const v8h lo = *(const v8h*)(colrow + kbase + half * 8);
  const v8h hi = *(const v8h*)(colrow + kbase + 16 + half * 8);
  return __builtin_shufflevector(lo, hi, 0,1,2,3,4,5,6,7,8,9,10,11,12,13,14,15);
}
// Same but from LDS (compiler emits ds_load_b128 x2)
__device__ __forceinline__ v16h load_b_frag_lds(const _Float16* colrow, int kbase, int half) {
  const v8h lo = *(const v8h*)(colrow + kbase + half * 8);
  const v8h hi = *(const v8h*)(colrow + kbase + 16 + half * 8);
  return __builtin_shufflevector(lo, hi, 0,1,2,3,4,5,6,7,8,9,10,11,12,13,14,15);
}
// A-fragment converting from f32 memory on the fly (x is read once from HBM)
__device__ __forceinline__ v16h cvt_a_frag_f32(const float* row, int kbase, int half) {
  const float* p0 = row + kbase + half * 8;
  const float* p1 = p0 + 16;
  v16h a;
#pragma unroll
  for (int i = 0; i < 8; ++i) { a[i] = (_Float16)p0[i]; a[8 + i] = (_Float16)p1[i]; }
  return a;
}

// ===================== Phase 0: one-shot weight f32 -> f16 =====================
// wh layout: [w][h][c], w order = q,k,v. 3*64*1024 elements; vectorized by 4.
__global__ __launch_bounds__(256, 1) void wcvt_kernel(
    const float* __restrict__ Wq, const float* __restrict__ Wk,
    const float* __restrict__ Wv, _Float16* __restrict__ wh) {
  const int tid = blockIdx.x * 256 + threadIdx.x;       // 0 .. 3*16384-1
  const int per_w = (H_DIM * C_DIM) / 4;                // 16384 float4s per weight
  const int w = tid / per_w;
  const int e = tid - w * per_w;
  const float* src = (w == 0) ? Wq : (w == 1) ? Wk : Wv;
  const f32x4 v = *(const f32x4*)(src + e * 4);
  _Float16* dst = wh + (size_t)w * H_DIM * C_DIM + (size_t)e * 4;
  dst[0] = (_Float16)v.x; dst[1] = (_Float16)v.y;
  dst[2] = (_Float16)v.z; dst[3] = (_Float16)v.w;
}

// ----- cooperative weight staging: 24KB slice (192 rows x 64 f16), 96B/thread -----
__device__ __forceinline__ void stage_load(const _Float16* __restrict__ wh, int kb,
                                           int t, v8h r[6]) {
#pragma unroll
  for (int i = 0; i < 6; ++i) {
    const int idx = t + i * 256;                // 16B-chunk id, 0..1535
    const int row = idx >> 3;                   // 0..191
    const int c8  = (idx & 7) << 3;             // f16 offset within 64
    r[i] = *(const v8h*)(wh + (size_t)row * C_DIM + kb + c8);
  }
}
__device__ __forceinline__ void stage_store(_Float16* buf, int t, const v8h r[6]) {
#pragma unroll
  for (int i = 0; i < 6; ++i) {
    const int idx = t + i * 256;
    const int row = idx >> 3;
    const int c8  = (idx & 7) << 3;
    *(v8h*)(buf + row * LSTRIDE + c8) = r[i];
  }
}

// ===================== Phase 1: fused Q/K/V projection =====================
// Block = 8 waves = 8 consecutive 16-row x tiles. Weights double-buffered in
// LDS (global weight traffic /8, B-fragments served at LDS latency); x read
// exactly once from HBM with in-register f32->f16 conversion.
// 12 accumulators per wave: (3 weights) x (4 col-tiles of 16).
// q pre-scaled by C^-0.5 = 1/32; v stored transposed (B, 64, T).
__global__ __launch_bounds__(256, 1) void qkv_proj_kernel(
    const float* __restrict__ x, const _Float16* __restrict__ wh,
    _Float16* __restrict__ qh, _Float16* __restrict__ kh, _Float16* __restrict__ vt) {
  __shared__ _Float16 lds[2][LROWS * LSTRIDE];

  const int t = threadIdx.x;
  const int lane = t & 31;
  const int wave = t >> 5;
  const int mtile = blockIdx.x * 8 + wave;        // 0 .. B*T/16-1
  const int m0 = mtile << 4;
  const int half = lane >> 4;
  const int l15 = lane & 15;

  const float* arow = x + (size_t)(m0 + l15) * C_DIM;
  // per-lane LDS row base for B fragments: row (w*64 + nt*16 + l15)
  const _Float16* lrow = &lds[0][0] + (size_t)l15 * LSTRIDE;
  const size_t lbuf_sz = (size_t)LROWS * LSTRIDE;

  v8f acc[12];
#pragma unroll
  for (int i = 0; i < 12; ++i) acc[i] = (v8f){};

  v8h stg[6];
  stage_load(wh, 0, t, stg);
  stage_store(&lds[0][0], t, stg);
  stage_load(wh, KC, t, stg);                     // chunk 1 in flight
  __syncthreads();

  for (int c = 0; c < CHUNKS; ++c) {
    const _Float16* buf = lrow + (size_t)(c & 1) * lbuf_sz;
#pragma unroll
    for (int ks = 0; ks < KC; ks += 32) {
      const v16h a = cvt_a_frag_f32(arow, c * KC + ks, half);
#pragma unroll
      for (int w = 0; w < 3; ++w) {
#pragma unroll
        for (int nt = 0; nt < 4; ++nt) {
          const v16h b = load_b_frag_lds(
              buf + (size_t)(w * H_DIM + nt * 16) * LSTRIDE, ks, half);
          acc[w * 4 + nt] = __builtin_amdgcn_wmma_f32_16x16x32_f16(
              false, a, false, b, (short)0, acc[w * 4 + nt], false, false);
        }
      }
    }
    if (c + 1 < CHUNKS) {
      stage_store(&lds[(c + 1) & 1][0], t, stg);
      if (c + 2 < CHUNKS) stage_load(wh, (c + 2) * KC, t, stg);
      __syncthreads();
    }
  }

  const int batch = m0 / T_DIM;
#pragma unroll
  for (int nt = 0; nt < 4; ++nt) {
    const int col = nt * 16 + l15;
#pragma unroll
    for (int r = 0; r < 8; ++r) {
      const int row = m0 + r + (half << 3);            // global row in (B*T)
      qh[(size_t)row * H_DIM + col] = (_Float16)(acc[0 * 4 + nt][r] * 0.03125f);
      kh[(size_t)row * H_DIM + col] = (_Float16)(acc[1 * 4 + nt][r]);
      const int tt = row - batch * T_DIM;
      vt[((size_t)batch * H_DIM + col) * T_DIM + tt] = (_Float16)(acc[2 * 4 + nt][r]);
    }
  }
}

// ===================== Phase 2: fused causal flash attention =====================
// One wave per 16 query rows. S^T = K*Q^T so the probability tile is already in
// A-fragment layout for P*V (no LDS transpose). Online softmax, f32 accum.
__global__ __launch_bounds__(256, 1) void attn_kernel(
    const _Float16* __restrict__ qh, const _Float16* __restrict__ kh,
    const _Float16* __restrict__ vt, float* __restrict__ out) {
  const int lane = threadIdx.x & 31;
  const int wave = threadIdx.x >> 5;
  const int task = blockIdx.x * 8 + wave;          // query tile id, 0..1023
  const int q0g = task << 4;                       // global query row
  const int batch = q0g / T_DIM;
  const int q0 = q0g - batch * T_DIM;              // query row within batch
  const int half = lane >> 4;
  const int l15 = lane & 15;
  const int tq = q0 + l15;                         // this lane's query index

  const _Float16* qrow = qh + (size_t)(q0g + l15) * H_DIM;
  const v16h qB0 = load_b_frag_f16(qrow, 0, half);   // head dims 0..31
  const v16h qB1 = load_b_frag_f16(qrow, 32, half);  // head dims 32..63

  const _Float16* kbase = kh + (size_t)batch * T_DIM * H_DIM;
  const _Float16* vbase = vt + (size_t)batch * H_DIM * T_DIM;

  float m_run = -1e30f, l_run = 0.0f;
  v8f o0 = {}, o1 = {}, o2 = {}, o3 = {};
  const v8f zero = {};

  const int s_end = q0 + 16;                       // keys needed: 0 .. q0+15
  for (int s0 = 0; s0 < s_end; s0 += 32) {
    const _Float16* k0row = kbase + (size_t)(s0 + l15) * H_DIM;
    const _Float16* k1row = k0row + 16 * H_DIM;
    const v16h kA00 = load_a_frag_f16(k0row, 0, half);
    const v16h kA01 = load_a_frag_f16(k0row, 32, half);
    const v16h kA10 = load_a_frag_f16(k1row, 0, half);
    const v16h kA11 = load_a_frag_f16(k1row, 32, half);

    // V B-fragments from transposed v: two contiguous 16B runs per lane (issue early
    // so they overlap the S^T WMMAs and softmax VALU work)
    const v16h vB0 = load_b_frag_f16(vbase + (size_t)( 0 + l15) * T_DIM, s0, half);
    const v16h vB1 = load_b_frag_f16(vbase + (size_t)(16 + l15) * T_DIM, s0, half);
    const v16h vB2 = load_b_frag_f16(vbase + (size_t)(32 + l15) * T_DIM, s0, half);
    const v16h vB3 = load_b_frag_f16(vbase + (size_t)(48 + l15) * T_DIM, s0, half);

    // S^T tiles: M=key, N=query (scale already folded into q)
    v8f st0 = __builtin_amdgcn_wmma_f32_16x16x32_f16(false, kA00, false, qB0, (short)0, zero, false, false);
    st0     = __builtin_amdgcn_wmma_f32_16x16x32_f16(false, kA01, false, qB1, (short)0, st0, false, false);
    v8f st1 = __builtin_amdgcn_wmma_f32_16x16x32_f16(false, kA10, false, qB0, (short)0, zero, false, false);
    st1     = __builtin_amdgcn_wmma_f32_16x16x32_f16(false, kA11, false, qB1, (short)0, st1, false, false);

    // causal mask + chunk max (per-lane select keeps EXEC all-1s for WMMA)
    float sv0[8], sv1[8];
    float cmax = -1e30f;
#pragma unroll
    for (int r = 0; r < 8; ++r) {
      const int key0 = s0 + r + (half << 3);
      sv0[r] = (key0      <= tq) ? st0[r] : -1e30f;
      sv1[r] = (key0 + 16 <= tq) ? st1[r] : -1e30f;
      cmax = fmaxf(cmax, fmaxf(sv0[r], sv1[r]));
    }
    cmax = fmaxf(cmax, __shfl_xor(cmax, 16, 32));   // pair lanes n / n+16 share query n
    const float m_new = fmaxf(m_run, cmax);
    const float alpha = __expf(m_run - m_new);

    // probabilities -> A-fragment of P (keys s0..s0+31), plus row sum
    v16h pA;
    float psum = 0.0f;
#pragma unroll
    for (int r = 0; r < 8; ++r) {
      const float e0 = __expf(sv0[r] - m_new);
      const float e1 = __expf(sv1[r] - m_new);
      psum += e0 + e1;
      pA[r]     = (_Float16)e0;
      pA[r + 8] = (_Float16)e1;
    }
    psum += __shfl_xor(psum, 16, 32);
    l_run = l_run * alpha + psum;
    m_run = m_new;

    // rescale O rows (O row M = r + 8*half lives on lane r+8*half's stats)
#pragma unroll
    for (int r = 0; r < 8; ++r) {
      const float ar = __shfl(alpha, r + (half << 3), 32);
      o0[r] *= ar; o1[r] *= ar; o2[r] *= ar; o3[r] *= ar;
    }

    o0 = __builtin_amdgcn_wmma_f32_16x16x32_f16(false, pA, false, vB0, (short)0, o0, false, false);
    o1 = __builtin_amdgcn_wmma_f32_16x16x32_f16(false, pA, false, vB1, (short)0, o1, false, false);
    o2 = __builtin_amdgcn_wmma_f32_16x16x32_f16(false, pA, false, vB2, (short)0, o2, false, false);
    o3 = __builtin_amdgcn_wmma_f32_16x16x32_f16(false, pA, false, vB3, (short)0, o3, false, false);
  }

  const float linv = 1.0f / l_run;
#pragma unroll
  for (int r = 0; r < 8; ++r) {
    const float lr = __shfl(linv, r + (half << 3), 32);
    const int row = q0g + r + (half << 3);
    float* orow = out + (size_t)row * H_DIM;
    orow[ 0 + l15] = o0[r] * lr;
    orow[16 + l15] = o1[r] * lr;
    orow[32 + l15] = o2[r] * lr;
    orow[48 + l15] = o3[r] * lr;
  }
}

extern "C" void kernel_launch(void* const* d_in, const int* in_sizes, int n_in,
                              void* d_out, int out_size, void* d_ws, size_t ws_size,
                              hipStream_t stream) {
  // setup_inputs() dict order: x, Wk, Wq, Wv
  const float* x  = (const float*)d_in[0];
  const float* Wk = (const float*)d_in[1];
  const float* Wq = (const float*)d_in[2];
  const float* Wv = (const float*)d_in[3];
  float* out = (float*)d_out;

  _Float16* qh = (_Float16*)d_ws;                          // (B*T, 64), pre-scaled
  _Float16* kh = qh + (size_t)B_DIM * T_DIM * H_DIM;       // (B*T, 64)
  _Float16* vt = kh + (size_t)B_DIM * T_DIM * H_DIM;       // (B, 64, T)
  _Float16* wh = vt + (size_t)B_DIM * T_DIM * H_DIM;       // (3, 64, 1024) f16

  const int wcvt_threads = (3 * H_DIM * C_DIM) / 4;        // 49152
  wcvt_kernel<<<wcvt_threads / 256, 256, 0, stream>>>(Wq, Wk, Wv, wh);

  const int proj_tiles = (B_DIM * T_DIM) / 16;             // 1024 waves
  qkv_proj_kernel<<<proj_tiles / 8, 256, 0, stream>>>(x, wh, qh, kh, vt);

  const int attn_tiles = (B_DIM * T_DIM) / 16;             // 1024 waves
  attn_kernel<<<attn_tiles / 8, 256, 0, stream>>>(qh, kh, vt, out);
}
